// Model_42709154791470
// MI455X (gfx1250) — compile-verified, matrix operand order checked by reading, not run on hardware
//
#include <hip/hip_runtime.h>
#include <cmath>

// ---------------------------------------------------------------------------
// AGCN forward for MI455X (gfx1250, wave32).  All conv GEMMs use
// v_wmma_f32_16x16x32_bf16 with f32 accumulation; activations stay fp32 in
// HBM and are converted to bf16 only while staging tiles into LDS.
// The model is HBM-bound (~5 GB traffic vs ~0.5 TFLOP), so staging uses
// b128 global loads + packed LDS stores on interior tiles, async global->LDS
// for the attention table, and global_prefetch for the next K chunk.
// ---------------------------------------------------------------------------

typedef __attribute__((ext_vector_type(16))) __bf16 v16bf;
typedef __attribute__((ext_vector_type(8)))  __bf16 v8bf;
typedef __attribute__((ext_vector_type(8)))  float  v8f;

#define NMB  16        // N*M person-sequences
#define VJ   25        // joints
#define NCLS 61        // NUM_CLASS + 1
#define FEAT 256       // final channels
#define BNS  0.9999950000374996f   // 1/sqrt(1 + 1e-5)

// float -> bf16 bits, round-to-nearest-even
__device__ __forceinline__ unsigned short f2bfu(float f) {
  unsigned u = __builtin_bit_cast(unsigned, f);
  u += 0x7FFFu + ((u >> 16) & 1u);
  return (unsigned short)(u >> 16);
}
__device__ __forceinline__ __bf16 f2bf(float f) {
  unsigned short s = f2bfu(f);
  return __builtin_bit_cast(__bf16, s);
}

// Build a 16x16x32 bf16 WMMA fragment from an LDS row (row-major over K).
// ISA layout: lanes 0-15 hold K = {0..7, 16..23}, lanes 16-31 hold
// K = {8..15, 24..31}; kb = 8*(lane>>4). Two aligned 16B LDS loads per frag.
__device__ __forceinline__ v16bf frag_ld(const __bf16* rowp, int kb) {
  v8bf lo = *(const v8bf*)(rowp + kb);
  v8bf hi = *(const v8bf*)(rowp + 16 + kb);
  v16bf r;
#pragma unroll
  for (int j = 0; j < 8; ++j) { r[j] = lo[j]; r[8 + j] = hi[j]; }
  return r;
}

// ---------------------------------------------------------------------------
// Generalized conv-as-GEMM.  OUT[n,o,to,v] = act( bn( bias( sum_{c,kt}
//   W[o, c*KT+kt] * X[n, c, to*stride+kt-pad, v] ) ) + resid )
// KT=1/stride=1 -> conv1x1;  KT=9,pad=4 -> temporal conv;  KT=1,stride=s ->
// strided residual 1x1.  flags: 1 = accumulate into OUT, 2 = ReLU.
// Tile: 32(O) x 64(cols); 4 waves, each wave owns two 16x16 WMMA tiles that
// share one B fragment.  K chunks of 32.
// ---------------------------------------------------------------------------
#define WG_TK 32
#define WG_PITCH (WG_TK + 8)

__global__ __launch_bounds__(128)
void wgemm_bf16(const float* __restrict__ Wt, const float* __restrict__ X,
                float* __restrict__ OUT,
                const float* __restrict__ bias,
                const float* __restrict__ bng, const float* __restrict__ bnb,
                const float* __restrict__ resid,
                int C, int Tin, int O, int Tout, int KT, int stride, int pad,
                int flags)
{
  __shared__ __bf16 As[32][WG_PITCH];
  __shared__ __bf16 Bs[64][WG_PITCH];

  const int n    = blockIdx.z;
  const int o0   = blockIdx.y * 32;
  const int col0 = blockIdx.x * 64;
  const int Ktot = C * KT;
  const int cols = Tout * VJ;
  const int tid  = threadIdx.x;
  const int lane = tid & 31;
  const int wv   = tid >> 5;
  const bool c11 = (KT == 1) && (stride == 1);   // contiguous B rows

  const float* Xn = X + (size_t)n * C * Tin * VJ;

  v8f acc0 = {}, acc1 = {};

  for (int k0 = 0; k0 < Ktot; k0 += WG_TK) {
    // Prefetch next K chunk of the B operand (speculative; safe if OOB).
    if (c11 && k0 + WG_TK < Ktot) {
      int kpf = k0 + WG_TK + (tid & 31);
      if (kpf < Ktot)
        __builtin_prefetch(&Xn[(size_t)kpf * cols + col0 + ((tid >> 5) << 4)], 0, 0);
    }

    // ---- stage A (weights) 32x32 ----
    if (o0 + 32 <= O && k0 + 32 <= Ktot && (Ktot & 3) == 0) {
#pragma unroll
      for (int it = 0; it < 2; ++it) {
        int u  = tid + it * 128;           // 0..255
        int r  = u >> 3, c4 = (u & 7) * 4;
        float4 w4 = *(const float4*)&Wt[(size_t)(o0 + r) * Ktot + k0 + c4];
        unsigned lo = ((unsigned)f2bfu(w4.y) << 16) | f2bfu(w4.x);
        unsigned hi = ((unsigned)f2bfu(w4.w) << 16) | f2bfu(w4.z);
        unsigned* dst = (unsigned*)&As[r][c4];
        dst[0] = lo; dst[1] = hi;
      }
    } else {
      for (int idx = tid; idx < 32 * WG_TK; idx += 128) {
        int r = idx >> 5, k = idx & 31;
        int o = o0 + r, kk = k0 + k;
        float v = 0.f;
        if (o < O && kk < Ktot) v = Wt[(size_t)o * Ktot + kk];
        As[r][k] = f2bf(v);
      }
    }
    // ---- stage B (activations), stored transposed: Bs[col][k] ----
    if (c11 && col0 + 64 <= cols && k0 + 32 <= Ktot && (cols & 3) == 0) {
#pragma unroll
      for (int it = 0; it < 4; ++it) {
        int u  = tid + it * 128;           // 0..511
        int k  = u >> 4, c4 = (u & 15) * 4;
        float4 x4 = *(const float4*)&Xn[(size_t)(k0 + k) * cols + col0 + c4];
        Bs[c4 + 0][k] = f2bf(x4.x);
        Bs[c4 + 1][k] = f2bf(x4.y);
        Bs[c4 + 2][k] = f2bf(x4.z);
        Bs[c4 + 3][k] = f2bf(x4.w);
      }
    } else {
      for (int idx = tid; idx < 64 * WG_TK; idx += 128) {
        int k = idx >> 6, cl = idx & 63;
        int kk = k0 + k, gc = col0 + cl;
        float v = 0.f;
        if (kk < Ktot && gc < cols) {
          int c  = kk / KT, kt = kk - c * KT;
          int to = gc / VJ, vv = gc - to * VJ;
          int ti = to * stride + kt - pad;
          if (ti >= 0 && ti < Tin)
            v = Xn[((size_t)c * Tin + ti) * VJ + vv];
        }
        Bs[cl][k] = f2bf(v);
      }
    }
    __syncthreads();
    int row = lane & 15, kb = (lane >> 4) * 8;
    v16bf b  = frag_ld(&Bs[wv * 16 + row][0], kb);
    v16bf a0 = frag_ld(&As[row][0], kb);
    v16bf a1 = frag_ld(&As[16 + row][0], kb);
    acc0 = __builtin_amdgcn_wmma_f32_16x16x32_bf16(false, a0, false, b,
                                                   (short)0, acc0, false, false);
    acc1 = __builtin_amdgcn_wmma_f32_16x16x32_bf16(false, a1, false, b,
                                                   (short)0, acc1, false, false);
    __syncthreads();
  }

  // Epilogue.  C/D layout: N = lane&15; VGPR r -> M = r + 8*(lane>>4).
  const int nn = lane & 15;
  const int mh = (lane >> 4) * 8;
  const int gc = col0 + wv * 16 + nn;
  if (gc < cols) {
    int to = gc / VJ, vv = gc - to * VJ;
#pragma unroll
    for (int half = 0; half < 2; ++half) {
      v8f av = half ? acc1 : acc0;
#pragma unroll
      for (int r = 0; r < 8; ++r) {
        int o = o0 + half * 16 + mh + r;
        if (o >= O) continue;
        float val = av[r];
        if (bias) val += bias[o];
        if (bng)  val = val * (bng[o] * BNS) + bnb[o];
        size_t oi = (((size_t)n * O + o) * Tout + to) * VJ + vv;
        if (resid)     val += resid[oi];
        if (flags & 1) val += OUT[oi];
        if (flags & 2) val = fmaxf(val, 0.f);
        OUT[oi] = val;
      }
    }
  }
}

// ---------------------------------------------------------------------------
// Attention Gram: L[n,i,u,w] += sum_k A1[n,k,u] * A2[n,k,w], k in this
// block's split-K slab.  25x25 output padded to 32x32 (2x2 WMMA tiles, one
// per wave), partial sums combined with global f32 atomics.
// ---------------------------------------------------------------------------
__global__ __launch_bounds__(128)
void gram_wmma(const float* __restrict__ A1, const float* __restrict__ A2,
               float* __restrict__ ATTL, int ic, int T, int isub, int kpers)
{
  __shared__ __bf16 As[32][WG_PITCH];
  __shared__ __bf16 Bs[32][WG_PITCH];
  const int n    = blockIdx.z;
  const int Ktot = ic * T;
  const int kbeg = blockIdx.x * kpers;
  const int kend = (kbeg + kpers < Ktot) ? (kbeg + kpers) : Ktot;
  const int tid  = threadIdx.x, lane = tid & 31, wv = tid >> 5;

  const float* A1n = A1 + (size_t)n * Ktot * VJ;
  const float* A2n = A2 + (size_t)n * Ktot * VJ;
  float* Ln = ATTL + ((size_t)n * 3 + isub) * (VJ * VJ);

  v8f acc = {};
  for (int k0 = kbeg; k0 < kend; k0 += WG_TK) {
    for (int idx = tid; idx < 32 * WG_TK; idx += 128) {
      int k = idx >> 5, m = idx & 31;
      int kk = k0 + k;
      float va = 0.f, vb = 0.f;
      if (m < VJ && kk < Ktot) {
        va = A1n[(size_t)kk * VJ + m];
        vb = A2n[(size_t)kk * VJ + m];
      }
      As[m][k] = f2bf(va);
      Bs[m][k] = f2bf(vb);
    }
    __syncthreads();
    int row = lane & 15, kb = (lane >> 4) * 8;
    int m0 = (wv & 1) * 16, q0 = (wv >> 1) * 16;
    v16bf a = frag_ld(&As[m0 + row][0], kb);
    v16bf b = frag_ld(&Bs[q0 + row][0], kb);
    acc = __builtin_amdgcn_wmma_f32_16x16x32_bf16(false, a, false, b,
                                                  (short)0, acc, false, false);
    __syncthreads();
  }
  int m0 = (wv & 1) * 16, q0 = (wv >> 1) * 16;
  int w  = q0 + (lane & 15);
  int mh = m0 + (lane >> 4) * 8;
  if (w < VJ) {
#pragma unroll
    for (int r = 0; r < 8; ++r) {
      int u = mh + r;
      if (u < VJ) atomicAdd(&Ln[u * VJ + w], acc[r]);
    }
  }
}

// softmax over rows (axis -2) per column, then + A[i] + PA[i]
__global__ void att_softmax(const float* __restrict__ L, float* __restrict__ ATT,
                            const float* __restrict__ Aadj,
                            const float* __restrict__ PA, float inv_k)
{
  int b = blockIdx.x;           // n*3 + i
  int i = b % 3;
  const float* Lb = L    + (size_t)b * (VJ * VJ);
  const float* At = Aadj + (size_t)i * (VJ * VJ);
  const float* Pi = PA   + (size_t)i * (VJ * VJ);
  float* Ob = ATT + (size_t)b * (VJ * VJ);
  int w = threadIdx.x;
  if (w >= VJ) return;
  float mx = -3.4e38f;
  for (int u = 0; u < VJ; ++u) mx = fmaxf(mx, Lb[u * VJ + w] * inv_k);
  float s = 0.f;
  for (int u = 0; u < VJ; ++u) s += expf(Lb[u * VJ + w] * inv_k - mx);
  float inv_s = 1.f / s;
  for (int u = 0; u < VJ; ++u)
    Ob[u * VJ + w] = expf(Lb[u * VJ + w] * inv_k - mx) * inv_s
                     + At[u * VJ + w] + Pi[u * VJ + w];
}

// z[n,(c,t),w] = sum_u x[n,(c,t),u] * att[n,i,u,w]
// The 625-float att table is brought into LDS with CDNA5 async global->LDS
// copies (ASYNCcnt-tracked), then consumed by all waves after a barrier.
__global__ __launch_bounds__(256)
void gcn_mix(const float* __restrict__ X, const float* __restrict__ ATT,
             float* __restrict__ Z, int rows, int isub)
{
  __shared__ float att[VJ * VJ];
  int n = blockIdx.z;
  const float* a = ATT + ((size_t)n * 3 + isub) * (VJ * VJ);
  for (int idx = threadIdx.x; idx < VJ * VJ; idx += 256) {
    unsigned           ldsa = (unsigned)(size_t)&att[idx];
    unsigned long long ga   = (unsigned long long)(size_t)&a[idx];
    asm volatile("global_load_async_to_lds_b32 %0, %1, off"
                 :: "v"(ldsa), "v"(ga) : "memory");
  }
  asm volatile("s_wait_asynccnt 0" ::: "memory");
  __syncthreads();
  int w  = threadIdx.x & 31;
  int rr = threadIdx.x >> 5;
  long row = (long)blockIdx.x * 8 + rr;
  if (row >= rows || w >= VJ) return;
  const float* xr = X + ((size_t)n * rows + row) * VJ;
  float s = 0.f;
#pragma unroll
  for (int u = 0; u < VJ; ++u) s += xr[u] * att[u * VJ + w];
  Z[((size_t)n * rows + row) * VJ + w] = s;
}

// h = relu(bn(y) + down)
__global__ void gcn_finalize(const float* __restrict__ Y,
                             const float* __restrict__ D,
                             float* __restrict__ H,
                             const float* __restrict__ g,
                             const float* __restrict__ b,
                             size_t total, int perC, int Cout)
{
  size_t i = (size_t)blockIdx.x * blockDim.x + threadIdx.x;
  if (i >= total) return;
  int o = (int)((i / perC) % Cout);
  float v = Y[i] * (g[o] * BNS) + b[o] + D[i];
  H[i] = fmaxf(v, 0.f);
}

// x[N,C,T,V,M] -> X0[(n*M+m), c, t, v]
__global__ void permute_in(const float* __restrict__ xin,
                           float* __restrict__ xo, size_t tot)
{
  size_t i = (size_t)blockIdx.x * blockDim.x + threadIdx.x;
  if (i >= tot) return;
  int v  = (int)(i % VJ);
  int t  = (int)((i / VJ) % 1024);
  int c  = (int)((i / ((size_t)VJ * 1024)) % 3);
  int nm = (int)(i / ((size_t)VJ * 1024 * 3));
  int n = nm >> 1, m = nm & 1;
  xo[i] = xin[((((size_t)n * 3 + c) * 1024 + t) * VJ + v) * 2 + m];
}

// emb[n,t,c] = mean over (m, v) of h[(2n+m), c, t, v]
__global__ void emb_reduce(const float* __restrict__ H, float* __restrict__ E,
                           int C, int T, size_t tot)
{
  size_t i = (size_t)blockIdx.x * blockDim.x + threadIdx.x;
  if (i >= tot) return;
  int c = (int)(i % C);
  int t = (int)((i / C) % T);
  int n = (int)(i / ((size_t)C * T));
  float s = 0.f;
  for (int m = 0; m < 2; ++m) {
    const float* hp = H + ((((size_t)(n * 2 + m) * C + c) * T + t) * VJ);
    for (int v = 0; v < VJ; ++v) s += hp[v];
  }
  E[i] = s * (1.f / 50.f);
}

// row-normalize anchors: na = ac / max(||ac||, 1e-12)
__global__ __launch_bounds__(256)
void rownorm(const float* __restrict__ ac, float* __restrict__ na, int D)
{
  __shared__ float red[256];
  int r = blockIdx.x, tid = threadIdx.x;
  float s = 0.f;
  for (int d = tid; d < D; d += 256) { float v = ac[(size_t)r * D + d]; s += v * v; }
  red[tid] = s; __syncthreads();
  for (int o = 128; o > 0; o >>= 1) {
    if (tid < o) red[tid] += red[tid + o];
    __syncthreads();
  }
  float inv = 1.f / fmaxf(sqrtf(red[0]), 1e-12f);
  for (int d = tid; d < D; d += 256) na[(size_t)r * D + d] = ac[(size_t)r * D + d] * inv;
}

// frm[b, cl] = SCALE * <emb_row/||emb_row||, na[cl]>
__global__ __launch_bounds__(64)
void frm_kernel(const float* __restrict__ E, const float* __restrict__ NA,
                float* __restrict__ F, int C, int NCv)
{
  __shared__ float es[FEAT];
  __shared__ float red[64];
  int b = blockIdx.x, tid = threadIdx.x;
  const float* e = E + (size_t)b * C;
  float s = 0.f;
  for (int d = tid; d < C; d += 64) { float v = e[d]; es[d] = v; s += v * v; }
  red[tid] = s; __syncthreads();
  for (int o = 32; o > 0; o >>= 1) {
    if (tid < o) red[tid] += red[tid + o];
    __syncthreads();
  }
  float inv = 1.f / fmaxf(sqrtf(red[0]), 1e-12f);
  for (int cl = tid; cl < NCv; cl += 64) {
    float d = 0.f;
    for (int k = 0; k < C; ++k) d += es[k] * NA[(size_t)cl * C + k];
    F[(size_t)b * NCv + cl] = d * inv * 5.0f;
  }
}

// per (n, cl): 3-temperature softmax aggregation over t, sigmoid
__global__ void vid_kernel(const float* __restrict__ F, float* __restrict__ P,
                           int T, int NCv)
{
  int idx = blockIdx.x * blockDim.x + threadIdx.x;
  if (idx >= 8 * NCv) return;
  int n = idx / NCv, cl = idx % NCv;
  const float* f = F + (size_t)n * T * NCv + cl;
  const float temps[3] = {1.f, 2.f, 5.f};
  float acc = 0.f;
  for (int q = 0; q < 3; ++q) {
    float tt = temps[q];
    float mx = -3.4e38f;
    for (int t = 0; t < T; ++t) mx = fmaxf(mx, f[(size_t)t * NCv] * tt);
    float se = 0.f, sw = 0.f;
    for (int t = 0; t < T; ++t) {
      float v = f[(size_t)t * NCv];
      float e = expf(v * tt - mx);
      se += e; sw += v * e;
    }
    acc += sw / se;
  }
  float vid = acc * (2.f / 3.f);
  P[idx] = 1.f / (1.f + expf(-vid));
}

// align_corners linear interpolation along t: [8,T,NC] -> [8,Tout,NC]
__global__ void interp_kernel(const float* __restrict__ F, float* __restrict__ O,
                              int T, int Tout, int NCv)
{
  size_t idx = (size_t)blockIdx.x * blockDim.x + threadIdx.x;
  size_t tot = (size_t)8 * Tout * NCv;
  if (idx >= tot) return;
  int cl = (int)(idx % NCv);
  int tp = (int)((idx / NCv) % Tout);
  int n  = (int)(idx / ((size_t)NCv * Tout));
  float pos = (float)tp * (float)(T - 1) / (float)(Tout - 1);
  int lo = (int)pos; if (lo > T - 1) lo = T - 1;
  int hi = (lo + 1 < T) ? lo + 1 : T - 1;
  float w = pos - (float)lo;
  O[idx] = F[((size_t)n * T + lo) * NCv + cl] * (1.f - w)
         + F[((size_t)n * T + hi) * NCv + cl] * w;
}

// ---------------------------------------------------------------------------
// Host orchestration
// ---------------------------------------------------------------------------
extern "C" void kernel_launch(void* const* d_in, const int* in_sizes, int n_in,
                              void* d_out, int out_size, void* d_ws, size_t ws_size,
                              hipStream_t stream)
{
  (void)in_sizes; (void)n_in; (void)out_size; (void)ws_size;
  const float* x_in = (const float*)d_in[0];
  const float* Aadj = (const float*)d_in[1];

  struct Cfg { int cin, cout, s; bool res; };
  const Cfg cfg[10] = {{3,64,1,false},{64,64,2,true},{64,64,1,true},{64,64,1,true},
                       {64,128,2,true},{128,128,1,true},{128,128,1,true},
                       {128,256,2,true},{256,256,1,true},{256,256,1,true}};

  struct LP {
    const float *Wa,*ba,*Wb,*bb,*Wd,*bd,*PA,*gg,*gb;
    const float *Wdn,*bdn,*dg,*db;
    const float *tW,*tb,*tg,*tbb;
    const float *rW,*rb,*rg,*rbb;
    bool down, resconv, res;
  } L[10];

  int p = 2;  // params flattened in dict-insertion order after x, A
  for (int l = 0; l < 10; ++l) {
    LP& P = L[l]; const Cfg& c = cfg[l];
    P.Wa = (const float*)d_in[p++]; P.ba = (const float*)d_in[p++];
    P.Wb = (const float*)d_in[p++]; P.bb = (const float*)d_in[p++];
    P.Wd = (const float*)d_in[p++]; P.bd = (const float*)d_in[p++];
    P.PA = (const float*)d_in[p++]; P.gg = (const float*)d_in[p++];
    P.gb = (const float*)d_in[p++];
    P.down = (c.cin != c.cout);
    P.Wdn = P.bdn = P.dg = P.db = nullptr;
    if (P.down) {
      P.Wdn = (const float*)d_in[p++]; P.bdn = (const float*)d_in[p++];
      P.dg  = (const float*)d_in[p++]; P.db  = (const float*)d_in[p++];
    }
    P.tW = (const float*)d_in[p++]; P.tb  = (const float*)d_in[p++];
    P.tg = (const float*)d_in[p++]; P.tbb = (const float*)d_in[p++];
    P.res = c.res;
    P.resconv = c.res && (c.cin != c.cout || c.s != 1);
    P.rW = P.rb = P.rg = P.rbb = nullptr;
    if (P.resconv) {
      P.rW = (const float*)d_in[p++]; P.rb  = (const float*)d_in[p++];
      P.rg = (const float*)d_in[p++]; P.rbb = (const float*)d_in[p++];
    }
  }
  const float* acs[2] = {(const float*)d_in[p], (const float*)d_in[p + 1]};

  // Workspace carve (floats)
  float* ws = (float*)d_ws;
  size_t off = 0;
  const size_t ACT = (size_t)NMB * 64 * 1024 * VJ;          // 26.2M floats max
  const size_t ASZ = (size_t)NMB * 16 * 1024 * VJ;          // a1/a2 max
  const size_t ATT_SZ = (size_t)NMB * 3 * VJ * VJ;
  float* B0   = ws + off; off += ACT;
  float* B1   = ws + off; off += ACT;
  float* Yb   = ws + off; off += ACT;
  float* Zb   = ws + off; off += ACT;
  float* Hb   = ws + off; off += ACT;
  float* DR   = ws + off; off += ACT;
  float* A1b  = ws + off; off += ASZ;
  float* A2b  = ws + off; off += ASZ;
  float* ATTL = ws + off; off += ATT_SZ;
  float* ATT  = ws + off; off += ATT_SZ;
  float* EMB  = ws + off; off += (size_t)8 * 128 * FEAT;
  float* NA   = ws + off; off += (size_t)NCLS * FEAT;
  float* FRM  = ws + off; off += (size_t)8 * 128 * NCLS;

  // x -> [16, 3, 1024, 25]
  {
    size_t tot = (size_t)NMB * 3 * 1024 * VJ;
    permute_in<<<dim3((unsigned)((tot + 255) / 256)), 256, 0, stream>>>(x_in, B0, tot);
  }

  float* cur = B0; float* nxt = B1;
  int Tcur = 1024;
  const dim3 blk(128);

  for (int l = 0; l < 10; ++l) {
    const Cfg& c = cfg[l]; LP& P = L[l];
    const int cin = c.cin, cout = c.cout, s = c.s;
    const int Tin = Tcur, Tout = (Tin - 1) / s + 1;
    const int ic = cout / 4;
    const int colsIn = Tin * VJ;

    hipMemsetAsync(ATTL, 0, ATT_SZ * sizeof(float), stream);

    // --- unit_gcn: attention logits per subset ---
    for (int i = 0; i < 3; ++i) {
      dim3 gA((colsIn + 63) / 64, (ic + 31) / 32, NMB);
      wgemm_bf16<<<gA, blk, 0, stream>>>(P.Wa + (size_t)i * ic * cin, cur, A1b,
          P.ba + i * ic, nullptr, nullptr, nullptr, cin, Tin, ic, Tin, 1, 1, 0, 0);
      wgemm_bf16<<<gA, blk, 0, stream>>>(P.Wb + (size_t)i * ic * cin, cur, A2b,
          P.bb + i * ic, nullptr, nullptr, nullptr, cin, Tin, ic, Tin, 1, 1, 0, 0);
      int Ktot = ic * Tin;
      int nslab = 32;
      int kpers = (((Ktot + nslab - 1) / nslab + 31) / 32) * 32;
      gram_wmma<<<dim3(nslab, 1, NMB), blk, 0, stream>>>(A1b, A2b, ATTL, ic, Tin, i, kpers);
    }
    att_softmax<<<dim3(NMB * 3), dim3(32), 0, stream>>>(ATTL, ATT, Aadj, P.PA,
                                                        1.f / (float)(ic * Tin));
    // --- graph mix + output conv, accumulated over subsets ---
    for (int i = 0; i < 3; ++i) {
      int rows = cin * Tin;
      gcn_mix<<<dim3((rows + 7) / 8, 1, NMB), dim3(256), 0, stream>>>(cur, ATT, Zb, rows, i);
      dim3 gD((colsIn + 63) / 64, (cout + 31) / 32, NMB);
      wgemm_bf16<<<gD, blk, 0, stream>>>(P.Wd + (size_t)i * cout * cin, Zb, Yb,
          P.bd + i * cout, nullptr, nullptr, nullptr, cin, Tin, cout, Tin, 1, 1, 0,
          (i ? 1 : 0));
    }
    // --- downsample branch + gcn finalize ---
    const float* dsrc = cur;
    if (P.down) {
      dim3 gDn((colsIn + 63) / 64, (cout + 31) / 32, NMB);
      wgemm_bf16<<<gDn, blk, 0, stream>>>(P.Wdn, cur, DR, P.bdn, P.dg, P.db,
          nullptr, cin, Tin, cout, Tin, 1, 1, 0, 0);
      dsrc = DR;
    }
    {
      size_t tot = (size_t)NMB * cout * Tin * VJ;
      gcn_finalize<<<dim3((unsigned)((tot + 255) / 256)), 256, 0, stream>>>(
          Yb, dsrc, Hb, P.gg, P.gb, tot, Tin * VJ, cout);
    }
    // --- residual branch ---
    const float* resid = nullptr;
    if (P.res) {
      if (P.resconv) {
        dim3 gR((Tout * VJ + 63) / 64, (cout + 31) / 32, NMB);
        wgemm_bf16<<<gR, blk, 0, stream>>>(P.rW, cur, DR, P.rb, P.rg, P.rbb,
            nullptr, cin, Tin, cout, Tout, 1, s, 0, 0);
        resid = DR;
      } else {
        resid = cur;
      }
    }
    // --- temporal conv 9x1 + bn + residual + relu ---
    {
      dim3 gT((Tout * VJ + 63) / 64, (cout + 31) / 32, NMB);
      wgemm_bf16<<<gT, blk, 0, stream>>>(P.tW, Hb, nxt, P.tb, P.tg, P.tbb,
          resid, cout, Tin, cout, Tout, 9, s, 4, 2);
    }
    float* t2 = cur; cur = nxt; nxt = t2;
    Tcur = Tout;
  }

  // --- classifier (twice: ac1, ac2) ---
  const int Tf = Tcur;   // 128
  {
    size_t tot = (size_t)8 * Tf * FEAT;
    emb_reduce<<<dim3((unsigned)((tot + 255) / 256)), 256, 0, stream>>>(cur, EMB, FEAT, Tf, tot);
  }
  float* outp = (float*)d_out;
  const size_t half = (size_t)8 * NCLS + (size_t)8 * 1024 * NCLS;
  for (int pass = 0; pass < 2; ++pass) {
    size_t base = pass ? half : 0;
    rownorm<<<dim3(NCLS), 256, 0, stream>>>(acs[pass], NA, FEAT);
    frm_kernel<<<dim3(8 * Tf), 64, 0, stream>>>(EMB, NA, FRM, FEAT, NCLS);
    vid_kernel<<<dim3(8), 64, 0, stream>>>(FRM, outp + base, Tf, NCLS);
    interp_kernel<<<dim3((8 * 1024 * NCLS + 255) / 256), 256, 0, stream>>>(
        FRM, outp + base + 8 * NCLS, Tf, 1024, NCLS);
  }
}
